// ParameterizedTransformer_10153302687982
// MI455X (gfx1250) — compile-verified, matrix operand order checked by reading, not run on hardware
//
#include <hip/hip_runtime.h>
#include <hip/hip_bf16.h>
#include <math.h>

// ---------------------------------------------------------------------------
// Graph TransformerConv x3 for MI455X (gfx1250, wave32, WMMA).
// Dense projections use v_wmma_f32_16x16x32_bf16 (f32 accum).
// Edge attention uses L2-resident gathers + float atomics for segment softmax.
// ---------------------------------------------------------------------------

typedef __attribute__((ext_vector_type(16))) __bf16 v16bf;
typedef __attribute__((ext_vector_type(8)))  float  v8f;

#define NODES 50000
#define EDGES 1600000
#define FDIM  128      // in_ch for every layer
#define D12   128      // heads*out_ch, layers 1-2
#define D3    16       // layer 3 output width

union BFrag { unsigned u[8]; v16bf v; };

static __device__ __forceinline__ unsigned short f32_to_bf16(float f) {
    unsigned u = __float_as_uint(f);
    u += 0x7FFFu + ((u >> 16) & 1u);        // round-to-nearest-even
    return (unsigned short)(u >> 16);
}

// ordered-uint encoding so unsigned atomicMax == float max
static __device__ __forceinline__ unsigned f2ord(float f) {
    unsigned u = __float_as_uint(f);
    return (u & 0x80000000u) ? ~u : (u | 0x80000000u);
}
static __device__ __forceinline__ float ord2f(unsigned u) {
    return __uint_as_float((u & 0x80000000u) ? (u ^ 0x80000000u) : ~u);
}

// ---- elementwise f32 -> bf16 -----------------------------------------------
__global__ void k_cvt_bf16(const float* __restrict__ src,
                           unsigned short* __restrict__ dst, int n) {
    int i = blockIdx.x * blockDim.x + threadIdx.x;
    if (i < n) dst[i] = f32_to_bf16(src[i]);
}

// ---- weight convert + transpose: W[K][D] f32 -> Wt[D][K] bf16 --------------
__global__ void k_cvt_wt(const float* __restrict__ W,
                         unsigned short* __restrict__ Wt, int K, int D) {
    int i = blockIdx.x * blockDim.x + threadIdx.x;
    if (i >= K * D) return;
    int n = i / K, k = i - n * K;
    Wt[i] = f32_to_bf16(W[k * D + n]);
}

// ---- WMMA GEMM: C[nRows][DCOLS] = A[nRows][128](bf16) @ W + bias -----------
// Block = 256 threads = 8 waves; each block does 128 rows, each wave 16 rows.
// A block-tile staged in LDS; Wt (tiny, L2-resident) read directly.
template <int DCOLS>
__global__ void __launch_bounds__(256)
k_gemm_bf16(const unsigned short* __restrict__ A,   // [nRows][128] bf16
            const unsigned short* __restrict__ Wt,  // [DCOLS][128] bf16 (transposed)
            const float* __restrict__ bias,         // [DCOLS]
            float* __restrict__ C,                  // [nRows][DCOLS]
            int nRows) {
    __shared__ unsigned aTile[128 * 64];            // 128 rows x 128 bf16 = 32KB
    const int tid = threadIdx.x;
    const int rowBase = blockIdx.x * 128;

    // cooperative stage of A tile (zero-fill guard rows)
    const uint4 zero4 = make_uint4(0u, 0u, 0u, 0u);
    #pragma unroll
    for (int it = 0; it < 8; ++it) {
        int c   = tid + it * 256;                   // 2048 uint4 chunks
        int r   = c >> 4;                           // 16 uint4 per row
        int seg = c & 15;
        int g   = rowBase + r;
        uint4 vv = zero4;
        if (g < nRows) vv = ((const uint4*)(A + (size_t)g * FDIM))[seg];
        ((uint4*)aTile)[c] = vv;
    }
    __syncthreads();

    const int wave = tid >> 5;
    const int lane = tid & 31;
    const int m    = lane & 15;
    const int hi   = lane >> 4;                     // lanes 16-31: K offset +8
    const int waveRow = wave * 16;

    v8f acc[DCOLS / 16] = {};
    const unsigned* __restrict__ WtU = (const unsigned*)Wt;

    #pragma unroll
    for (int kt = 0; kt < FDIM / 32; ++kt) {
        BFrag af;
        #pragma unroll
        for (int d = 0; d < 8; ++d) {
            // 16-bit A layout: vgpr d holds K pair 2*(d&3) (+16 for d>=4), +8 for hi lanes
            int kOff = 2 * (d & 3) + ((d >= 4) ? 16 : 0) + (hi ? 8 : 0);
            int K = kt * 32 + kOff;
            af.u[d] = aTile[(waveRow + m) * 64 + (K >> 1)];
        }
        #pragma unroll
        for (int ct = 0; ct < DCOLS / 16; ++ct) {
            BFrag bfm;
            int n = ct * 16 + m;                    // B: lane holds column n
            #pragma unroll
            for (int d = 0; d < 8; ++d) {
                int kOff = 2 * (d & 3) + ((d >= 4) ? 16 : 0) + (hi ? 8 : 0);
                int K = kt * 32 + kOff;
                bfm.u[d] = WtU[n * 64 + (K >> 1)];
            }
            acc[ct] = __builtin_amdgcn_wmma_f32_16x16x32_bf16(
                false, af.v, false, bfm.v, (short)0, acc[ct], false, false);
        }
    }

    // C/D layout: vgpr r -> row (r + 8*hi), col = lane&15
    #pragma unroll
    for (int ct = 0; ct < DCOLS / 16; ++ct) {
        int n = ct * 16 + m;
        float b = bias[n];
        #pragma unroll
        for (int r = 0; r < 8; ++r) {
            int gr = rowBase + waveRow + hi * 8 + r;
            if (gr < nRows)
                C[(size_t)gr * DCOLS + n] = acc[ct][r] + b;
        }
    }
}

// ---- attention stage --------------------------------------------------------
__global__ void k_attn_init(float* __restrict__ acc, float* __restrict__ s,
                            unsigned* __restrict__ m, int nAcc, int nH,
                            unsigned minit) {
    int i = blockIdx.x * blockDim.x + threadIdx.x;
    if (i < nAcc) acc[i] = 0.f;
    if (i < nH) { s[i] = 0.f; m[i] = minit; }
}

// pass 1: alpha[e,h] = <q[dst], k[src]>_h * scale ; segment max via atomicMax
__global__ void k_attn_logits(const int* __restrict__ src,
                              const int* __restrict__ dst,
                              const float* __restrict__ q,
                              const float* __restrict__ k,
                              float* __restrict__ alpha,
                              unsigned* __restrict__ mmax,
                              int H, int Cc, int D, float scale) {
    int i = blockIdx.x * blockDim.x + threadIdx.x;
    if (i >= EDGES * H) return;
    int e = i / H, h = i - e * H;
    int sN = src[e], dN = dst[e];
    const float4* qr = (const float4*)(q + (size_t)dN * D + h * Cc);
    const float4* kr = (const float4*)(k + (size_t)sN * D + h * Cc);
    float acc = 0.f;
    for (int c = 0; c < Cc / 4; ++c) {
        float4 a = qr[c], b = kr[c];
        acc += a.x * b.x + a.y * b.y + a.z * b.z + a.w * b.w;
    }
    float al = acc * scale;
    alpha[i] = al;
    atomicMax(&mmax[dN * H + h], f2ord(al));
}

// pass 2: a = exp(alpha - m[dst]); s[dst] += a; acc[dst] += a * v[src]
__global__ void k_attn_scatter(const int* __restrict__ src,
                               const int* __restrict__ dst,
                               const float* __restrict__ v,
                               const float* __restrict__ alpha,
                               const unsigned* __restrict__ mmax,
                               float* __restrict__ ssum,
                               float* __restrict__ acc,
                               int H, int Cc, int D) {
    int i = blockIdx.x * blockDim.x + threadIdx.x;
    if (i >= EDGES * H) return;
    int e = i / H, h = i - e * H;
    int sN = src[e], dN = dst[e];
    float a = __expf(alpha[i] - ord2f(mmax[dN * H + h]));
    atomicAdd(&ssum[dN * H + h], a);
    const float* vr = v + (size_t)sN * D + h * Cc;
    float*       ar = acc + (size_t)dN * D + h * Cc;
    for (int c = 0; c < Cc; ++c)
        atomicAdd(&ar[c], a * vr[c]);
}

// pass 3: out = acc / (s + eps) + skip ; optional ReLU
__global__ void k_attn_final(const float* __restrict__ acc,
                             const float* __restrict__ ssum,
                             const float* __restrict__ skip,
                             float* __restrict__ out,
                             int H, int Cc, int D, int relu, int nTot) {
    int i = blockIdx.x * blockDim.x + threadIdx.x;
    if (i >= nTot) return;
    int n = i / D, dcol = i - n * D;
    int h = dcol / Cc;
    float val = acc[i] / (ssum[n * H + h] + 1e-16f) + skip[i];
    if (relu) val = fmaxf(val, 0.f);
    out[i] = val;
}

// ---------------------------------------------------------------------------
extern "C" void kernel_launch(void* const* d_in, const int* in_sizes, int n_in,
                              void* d_out, int out_size, void* d_ws, size_t ws_size,
                              hipStream_t stream) {
    (void)in_sizes; (void)n_in; (void)out_size; (void)ws_size;

    const float* x    = (const float*)d_in[0];
    const int*   edge = (const int*)d_in[1];     // [2][EDGES]
    const int*   src  = edge;
    const int*   dst  = edge + EDGES;
    // params per layer at d_in[2+8l+...]: Wq,bq,Wk,bk,Wv,bv,Ws,bs

    // ---- workspace carve-out (~170 MB) ----
    char*  ws  = (char*)d_ws;
    size_t off = 0;
    auto alloc = [&](size_t bytes) -> void* {
        void* p = ws + off;
        off += (bytes + 255) & ~(size_t)255;
        return p;
    };
    unsigned short* wt[3][4];
    for (int l = 0; l < 3; ++l) {
        int Dd = (l == 2) ? D3 : D12;
        for (int j = 0; j < 4; ++j)
            wt[l][j] = (unsigned short*)alloc((size_t)FDIM * Dd * 2);
    }
    unsigned short* xbf  = (unsigned short*)alloc((size_t)NODES * FDIM * 2);
    float* Q    = (float*)alloc((size_t)NODES * D12 * 4);   // doubles as h buffer
    float* K    = (float*)alloc((size_t)NODES * D12 * 4);
    float* V    = (float*)alloc((size_t)NODES * D12 * 4);
    float* SKIP = (float*)alloc((size_t)NODES * D12 * 4);
    float* ACC  = (float*)alloc((size_t)NODES * D12 * 4);
    float* ALPH = (float*)alloc((size_t)EDGES * 4 * 4);
    float* SSUM = (float*)alloc((size_t)NODES * 4 * 4);
    unsigned* MMAX = (unsigned*)alloc((size_t)NODES * 4 * 4);

    const int T = 256;
    const unsigned MINIT = 0x00800000u;          // f2ord(-FLT_MAX)

    // ---- convert all weights once (transposed, bf16) ----
    for (int l = 0; l < 3; ++l) {
        int Dd = (l == 2) ? D3 : D12;
        for (int j = 0; j < 4; ++j) {
            const float* W = (const float*)d_in[2 + 8 * l + 2 * j];
            k_cvt_wt<<<(FDIM * Dd + T - 1) / T, T, 0, stream>>>(W, wt[l][j], FDIM, Dd);
        }
    }
    // ---- x -> bf16 ----
    k_cvt_bf16<<<(NODES * FDIM + T - 1) / T, T, 0, stream>>>(x, xbf, NODES * FDIM);

    const int gemmGrid = (NODES + 127) / 128;    // 391 blocks

    for (int l = 0; l < 3; ++l) {
        const int Dd = (l == 2) ? D3 : D12;
        const int H  = (l == 2) ? 1 : 4;
        const int Cc = (l == 2) ? 16 : 32;
        const float scale = 1.0f / sqrtf((float)Cc);
        const int relu = (l < 2) ? 1 : 0;
        float* dest = (l == 2) ? (float*)d_out : Q;

        if (l > 0)   // previous layer output lives in Q; convert before gemm clobbers it
            k_cvt_bf16<<<(NODES * FDIM + T - 1) / T, T, 0, stream>>>(Q, xbf, NODES * FDIM);

        float* outs[4] = { Q, K, V, SKIP };
        for (int j = 0; j < 4; ++j) {
            const float* bias = (const float*)d_in[2 + 8 * l + 2 * j + 1];
            if (Dd == D12)
                k_gemm_bf16<D12><<<gemmGrid, 256, 0, stream>>>(xbf, wt[l][j], bias, outs[j], NODES);
            else
                k_gemm_bf16<D3><<<gemmGrid, 256, 0, stream>>>(xbf, wt[l][j], bias, outs[j], NODES);
        }

        k_attn_init<<<(NODES * Dd + T - 1) / T, T, 0, stream>>>(
            ACC, SSUM, MMAX, NODES * Dd, NODES * H, MINIT);

        int eThreads = EDGES * H;
        k_attn_logits<<<(eThreads + T - 1) / T, T, 0, stream>>>(
            src, dst, Q, K, ALPH, MMAX, H, Cc, Dd, scale);
        k_attn_scatter<<<(eThreads + T - 1) / T, T, 0, stream>>>(
            src, dst, V, ALPH, MMAX, SSUM, ACC, H, Cc, Dd);
        k_attn_final<<<(NODES * Dd + T - 1) / T, T, 0, stream>>>(
            ACC, SSUM, SKIP, dest, H, Cc, Dd, relu, NODES * Dd);
    }
}
